// AutoEncoder_42743514530402
// MI455X (gfx1250) — matrix operand order repaired and checked by reading, not compiled, not run on hardware
//
#include <hip/hip_runtime.h>
#include <hip/hip_bf16.h>
#include <math.h>

typedef __bf16 bf16_t;
typedef __attribute__((ext_vector_type(16))) __bf16 v16bf;
typedef __attribute__((ext_vector_type(8)))  __bf16 v8bf;
typedef __attribute__((ext_vector_type(4)))  __bf16 v4bf;
typedef __attribute__((ext_vector_type(8)))  float  v8f;

#define CB    2048      // n_batches
#define CN    65536     // n points
#define CDIM  256
#define CHID  512
#define CMAXN 64
#define CMIDK 288
#define CMIDV 384
#define CMIDD 384
#define CMIDS 256
#define XOUT_ELEMS (CB * CMAXN * CDIM)   // 33554432

// ---------------------------------------------------------------- helpers ---

__device__ __forceinline__ float mishf(float v) {
    float sp = (v > 20.f) ? v : log1pf(expf(v));
    return v * tanhf(sp);
}

__device__ __forceinline__ v16bf cat8(v8bf lo, v8bf hi) {
    return __builtin_shufflevector(lo, hi, 0, 1, 2, 3, 4, 5, 6, 7,
                                           8, 9, 10, 11, 12, 13, 14, 15);
}

// A fragment (16x32) from row-major bf16 (global or LDS): two b128 loads.
// 16-bit A layout: lane half 0 -> K {0..7, 16..23}, half 1 -> K {8..15, 24..31}
__device__ __forceinline__ v16bf a_frag_bf16(const bf16_t* Arow0, int lda,
                                             int k0, int lane) {
    int m = lane & 15, half = lane >> 4;
    const bf16_t* r = Arow0 + (size_t)m * lda + k0 + half * 8;
    v8bf lo = *(const v8bf*)(r);
    v8bf hi = *(const v8bf*)(r + 16);
    return cat8(lo, hi);
}

// B fragment from pre-swizzled weights: one 32B contiguous read per lane
__device__ __forceinline__ v16bf b_frag_swz(const bf16_t* WS, int KT,
                                            int nt, int kt, int lane) {
    const v8bf* p = (const v8bf*)(WS + ((size_t)(nt * KT + kt) * 32 + lane) * 16);
    return cat8(p[0], p[1]);
}

#define WMMA_BF16(acc, a, b) \
    __builtin_amdgcn_wmma_f32_16x16x32_bf16(false, (a), false, (b), (short)0, (acc), false, false)

// ---------------------------------------------------------- prep kernels ----

__global__ void f32_to_bf16_kernel(const float* __restrict__ src,
                                   bf16_t* __restrict__ dst, int n) {
    int e = (blockIdx.x * blockDim.x + threadIdx.x) * 4;
    if (e < n) {
        v4bf o;
#pragma unroll
        for (int j = 0; j < 4; ++j) o[j] = (bf16_t)src[e + j];
        *(v4bf*)(dst + e) = o;
    }
}

// swizzle row-major f32 weight W[K][Nn] into fragment-major bf16:
// WS[((nt*KT + kt)*32 + lane)*16 + i] = W[kt*32 + (lane>>4)*8 + (i&7) + (i>>3)*16][nt*16 + (lane&15)]
__global__ void swz_kernel(const float* __restrict__ W, bf16_t* __restrict__ WS,
                           int K, int Nn) {
    int idx = blockIdx.x * blockDim.x + threadIdx.x;
    int KT = K >> 5, NT = Nn >> 4;
    if (idx >= NT * KT * 32) return;
    int lane = idx & 31;
    int kt = (idx >> 5) % KT;
    int nt = (idx >> 5) / KT;
    int col = lane & 15, half = lane >> 4;
    const float* p = W + (size_t)(kt * 32 + half * 8) * Nn + nt * 16 + col;
    bf16_t* o = WS + (size_t)idx * 16;
#pragma unroll
    for (int i = 0; i < 16; ++i) o[i] = (bf16_t)p[(size_t)(i + (i >> 3) * 8) * Nn];
}

// ------------------------------------------------------------ small kernels -

__global__ void zero_i32_kernel(int* p, int n) {
    int i = blockIdx.x * blockDim.x + threadIdx.x;
    if (i < n) p[i] = 0;
}

__global__ void count_kernel(const int* __restrict__ batch, int* cnt) {
    int i = blockIdx.x * blockDim.x + threadIdx.x;
    if (i < CN) atomicAdd(&cnt[batch[i]], 1);
}

__global__ void scan_kernel(const int* __restrict__ cnt, int* starts) {
    int run = 0;
    for (int b = 0; b < CB; ++b) { starts[b] = run; run += cnt[b]; }
}

__global__ void mag_kernel(const float* __restrict__ x, const float* __restrict__ rW,
                           const float* __restrict__ rb, float* __restrict__ mag) {
    int row  = blockIdx.x * 8 + (threadIdx.x >> 5);
    int lane = threadIdx.x & 31;
    const float* xr = x + (size_t)row * CDIM;
    float s = 0.f;
    for (int k = lane; k < CDIM; k += 32) s += xr[k] * rW[k];
#pragma unroll
    for (int off = 16; off; off >>= 1) s += __shfl_xor(s, off, 32);
    if (lane == 0) mag[row] = fabsf(s + rb[0]);
}

__global__ void rank_kernel(const int* __restrict__ cnt, const int* __restrict__ starts,
                            const float* __restrict__ mag, int* __restrict__ rank) {
    int b = blockIdx.x;
    int n = cnt[b], s = starts[b];
    for (int i = threadIdx.x; i < n; i += blockDim.x) {
        float mi = mag[s + i];
        int r = 0;
        for (int j = 0; j < n; ++j) {
            float mj = mag[s + j];
            r += (mj < mi) || (mj == mi && j < i);
        }
        rank[s + i] = (r < CMAXN) ? r : CMAXN;
    }
}

__global__ void keytab_kernel(const float* __restrict__ W1, const float* __restrict__ b1,
                              const float* __restrict__ g,  const float* __restrict__ be,
                              const float* __restrict__ W2, const float* __restrict__ b2,
                              float* __restrict__ tab) {
    __shared__ float h[CMIDK];
    __shared__ float red[2];
    int p = blockIdx.x, tid = threadIdx.x;
    for (int j = tid; j < CMIDK; j += blockDim.x)
        h[j] = ((p < CMAXN) ? W1[p * CMIDK + j] : 0.f) + b1[j];
    __syncthreads();
    if (tid == 0) {
        float m = 0.f;
        for (int j = 0; j < CMIDK; ++j) m += h[j];
        m /= CMIDK;
        float v = 0.f;
        for (int j = 0; j < CMIDK; ++j) { float d = h[j] - m; v += d * d; }
        v /= CMIDK;
        red[0] = m; red[1] = rsqrtf(v + 1e-5f);
    }
    __syncthreads();
    for (int j = tid; j < CMIDK; j += blockDim.x)
        h[j] = mishf((h[j] - red[0]) * red[1] * g[j] + be[j]);
    __syncthreads();
    for (int n = tid; n < CHID; n += blockDim.x) {
        float acc = b2[n];
        for (int j = 0; j < CMIDK; ++j) acc += h[j] * W2[j * CHID + n];
        tab[(size_t)p * CHID + n] = acc;
    }
}

__global__ void zinit_kernel(const int* __restrict__ cnt, const float* __restrict__ cW,
                             const float* __restrict__ cb, float* __restrict__ z) {
    int b = blockIdx.x;
    float nf = (float)cnt[b];
    for (int n = threadIdx.x; n < CHID; n += blockDim.x)
        z[(size_t)b * CHID + n] = nf * cW[n] + cb[n];
}

// -------------------------------------------------- fused encoder val-MLP ---
// 64-row tile, 4 waves x 16 rows; everything wave-local on Hb -> no barriers.
// N-tile loop unrolled x2: two independent WMMA accumulator chains.
__global__ __launch_bounds__(128) void encoder_mlp_kernel(
    const bf16_t* __restrict__ xb, const int* __restrict__ batch,
    const int* __restrict__ rank,
    const bf16_t* __restrict__ WS1, const float* __restrict__ b1,
    const float* __restrict__ g,  const float* __restrict__ be,
    const bf16_t* __restrict__ WS2, const float* __restrict__ b2,
    const float* __restrict__ ekey, float* __restrict__ z) {
    __shared__ bf16_t Hb[64 * CMIDV];      // 48 KB
    int tid = threadIdx.x, lane = tid & 31, wave = tid >> 5;
    int row0 = blockIdx.x * 64;
    int srow = wave * 16;
    int half = lane >> 4, n_lo = lane & 15;

    // Stage 1: H = x @ W1 + b1    (K = 256 -> 8 cached A fragments)
    v16bf Af[8];
#pragma unroll
    for (int kt = 0; kt < 8; ++kt)
        Af[kt] = a_frag_bf16(xb + (size_t)(row0 + srow) * CDIM, CDIM, kt * 32, lane);

    for (int nt = 0; nt < CMIDV / 16; nt += 2) {
        v8f acc0 = {}, acc1 = {};
#pragma unroll
        for (int kt = 0; kt < 8; ++kt) {
            acc0 = WMMA_BF16(acc0, Af[kt], b_frag_swz(WS1, 8, nt,     kt, lane));
            acc1 = WMMA_BF16(acc1, Af[kt], b_frag_swz(WS1, 8, nt + 1, kt, lane));
        }
        int n0 = nt * 16 + n_lo;
        float bb0 = b1[n0], bb1 = b1[n0 + 16];
#pragma unroll
        for (int r = 0; r < 8; ++r) {
            int m = srow + half * 8 + r;
            Hb[m * CMIDV + n0]      = (bf16_t)(acc0[r] + bb0);
            Hb[m * CMIDV + n0 + 16] = (bf16_t)(acc1[r] + bb1);
        }
    }

    // wave-parallel LayerNorm + mish over this wave's 16 rows
    for (int rr = 0; rr < 16; ++rr) {
        bf16_t* hr = Hb + (srow + rr) * CMIDV;
        float s = 0.f;
        for (int j = lane; j < CMIDV; j += 32) s += (float)hr[j];
#pragma unroll
        for (int off = 16; off; off >>= 1) s += __shfl_xor(s, off, 32);
        float mean = s * (1.f / CMIDV);
        float v = 0.f;
        for (int j = lane; j < CMIDV; j += 32) {
            float d = (float)hr[j] - mean;
            v += d * d;
        }
#pragma unroll
        for (int off = 16; off; off >>= 1) v += __shfl_xor(v, off, 32);
        float inv = rsqrtf(v * (1.f / CMIDV) + 1e-5f);
        for (int j = lane; j < CMIDV; j += 32) {
            float t = ((float)hr[j] - mean) * inv * g[j] + be[j];
            hr[j] = (bf16_t)mishf(t);
        }
    }

    int rk[8], bt[8];
#pragma unroll
    for (int r = 0; r < 8; ++r) {
        int row = row0 + srow + half * 8 + r;
        rk[r] = rank[row];
        bt[r] = batch[row];
    }

    // Stage 2: V = H @ W2 + b2 ; y = V * ekey[rank] ; z[batch] += y
    v16bf Hf[12];
#pragma unroll
    for (int kt = 0; kt < 12; ++kt)
        Hf[kt] = a_frag_bf16(Hb + srow * CMIDV, CMIDV, kt * 32, lane);

    for (int nt = 0; nt < CHID / 16; nt += 2) {
        v8f acc0 = {}, acc1 = {};
#pragma unroll
        for (int kt = 0; kt < 12; ++kt) {
            acc0 = WMMA_BF16(acc0, Hf[kt], b_frag_swz(WS2, 12, nt,     kt, lane));
            acc1 = WMMA_BF16(acc1, Hf[kt], b_frag_swz(WS2, 12, nt + 1, kt, lane));
        }
        int n0 = nt * 16 + n_lo;
        float bb0 = b2[n0], bb1 = b2[n0 + 16];
#pragma unroll
        for (int r = 0; r < 8; ++r) {
            const float* ek = ekey + (size_t)rk[r] * CHID + n0;
            float* zr = z + (size_t)bt[r] * CHID + n0;
            unsafeAtomicAdd(zr,      (acc0[r] + bb0) * ek[0]);
            unsafeAtomicAdd(zr + 16, (acc1[r] + bb1) * ek[16]);
        }
    }
}

// --------------------------------------------------------- size head MLP ---
__global__ __launch_bounds__(256) void size_kernel(
    const float* __restrict__ z, const float* __restrict__ W1, const float* __restrict__ b1,
    const float* __restrict__ g, const float* __restrict__ be,
    const float* __restrict__ W2, const float* __restrict__ b2, int* __restrict__ ndec) {
    __shared__ float h[CMIDS];
    __shared__ float red[2];
    __shared__ float dot[256];
    int b = blockIdx.x, tid = threadIdx.x;
    const float* zr = z + (size_t)b * CHID;
    float acc = b1[tid];
    for (int k = 0; k < CHID; ++k) acc += zr[k] * W1[k * CMIDS + tid];
    h[tid] = acc;
    __syncthreads();
    if (tid == 0) {
        float m = 0.f;
        for (int j = 0; j < CMIDS; ++j) m += h[j];
        m /= CMIDS;
        float v = 0.f;
        for (int j = 0; j < CMIDS; ++j) { float d = h[j] - m; v += d * d; }
        v /= CMIDS;
        red[0] = m; red[1] = rsqrtf(v + 1e-5f);
    }
    __syncthreads();
    dot[tid] = mishf((h[tid] - red[0]) * red[1] * g[tid] + be[tid]) * W2[tid];
    __syncthreads();
    for (int off = 128; off; off >>= 1) {
        if (tid < off) dot[tid] += dot[tid + off];
        __syncthreads();
    }
    if (tid == 0) {
        float logit = dot[0] + b2[0];
        float r = fminf(fmaxf(roundf(logit), 0.f), (float)CMAXN);
        ndec[b] = (int)r;
    }
}

// ------------------------------------------------------ fused decoder MLP ---
// one block per batch (64 positions); zp staged to LDS bf16 (one barrier),
// then each wave works its own 16-row strip; 2 independent WMMA chains.
__global__ __launch_bounds__(128) void decoder_mlp_kernel(
    const float* __restrict__ z, const float* __restrict__ dkey,
    const bf16_t* __restrict__ WS1, const float* __restrict__ b1,
    const bf16_t* __restrict__ WS2, const float* __restrict__ b2,
    const int* __restrict__ ndec, float* __restrict__ xout) {
    __shared__ bf16_t Zp[64 * CHID];       // 64 KB
    __shared__ bf16_t Hb[64 * CMIDD];      // 48 KB
    int tid = threadIdx.x, lane = tid & 31, wave = tid >> 5;
    int b = blockIdx.x;
    int srow = wave * 16;
    int half = lane >> 4, n_lo = lane & 15;
    const float* zrow = z + (size_t)b * CHID;
    int nd = ndec[b];

    // Stage 0: Zp[p][k] = bf16( z[k] * dkey[p][k] )
    for (int e = tid * 4; e < 64 * CHID; e += 128 * 4) {
        int p = e >> 9, k = e & (CHID - 1);
        const float* kd = dkey + (size_t)p * CHID + k;
        v4bf o;
#pragma unroll
        for (int j = 0; j < 4; ++j) o[j] = (bf16_t)(zrow[k + j] * kd[j]);
        *(v4bf*)(Zp + e) = o;
    }
    __syncthreads();

    // Stage 1: H = mish( Zp @ W1 + b1 )   (K = 512 -> 16 cached A fragments)
    {
        v16bf Af[16];
#pragma unroll
        for (int kt = 0; kt < 16; ++kt)
            Af[kt] = a_frag_bf16(Zp + srow * CHID, CHID, kt * 32, lane);

        for (int nt = 0; nt < CMIDD / 16; nt += 2) {
            v8f acc0 = {}, acc1 = {};
#pragma unroll
            for (int kt = 0; kt < 16; ++kt) {
                acc0 = WMMA_BF16(acc0, Af[kt], b_frag_swz(WS1, 16, nt,     kt, lane));
                acc1 = WMMA_BF16(acc1, Af[kt], b_frag_swz(WS1, 16, nt + 1, kt, lane));
            }
            int n0 = nt * 16 + n_lo;
            float bb0 = b1[n0], bb1 = b1[n0 + 16];
#pragma unroll
            for (int r = 0; r < 8; ++r) {
                int m = srow + half * 8 + r;
                Hb[m * CMIDD + n0]      = (bf16_t)mishf(acc0[r] + bb0);
                Hb[m * CMIDD + n0 + 16] = (bf16_t)mishf(acc1[r] + bb1);
            }
        }
    }

    // Stage 2: X = H @ W2 + b2, masked by position < n_dec (wave-local strip)
    v16bf Hf[12];
#pragma unroll
    for (int kt = 0; kt < 12; ++kt)
        Hf[kt] = a_frag_bf16(Hb + srow * CMIDD, CMIDD, kt * 32, lane);

    for (int nt = 0; nt < CDIM / 16; nt += 2) {
        v8f acc0 = {}, acc1 = {};
#pragma unroll
        for (int kt = 0; kt < 12; ++kt) {
            acc0 = WMMA_BF16(acc0, Hf[kt], b_frag_swz(WS2, 12, nt,     kt, lane));
            acc1 = WMMA_BF16(acc1, Hf[kt], b_frag_swz(WS2, 12, nt + 1, kt, lane));
        }
        int n0 = nt * 16 + n_lo;
        float bb0 = b2[n0], bb1 = b2[n0 + 16];
#pragma unroll
        for (int r = 0; r < 8; ++r) {
            int p = srow + half * 8 + r;
            bool on = p < nd;
            float* o = xout + ((size_t)b * CMAXN + p) * CDIM + n0;
            o[0]  = on ? (acc0[r] + bb0) : 0.f;
            o[16] = on ? (acc1[r] + bb1) : 0.f;
        }
    }
}

// ----------------------------------------------------- batch_out and mask ---
__global__ void aux_kernel(const int* __restrict__ ndec, float* __restrict__ out) {
    int i = blockIdx.x * blockDim.x + threadIdx.x;
    if (i < CB * CMAXN) {
        int b = i >> 6, p = i & 63;
        out[(size_t)XOUT_ELEMS + i] = (float)b;
        out[(size_t)XOUT_ELEMS + CB * CMAXN + i] = (p < ndec[b]) ? 1.f : 0.f;
    }
}

// ------------------------------------------------------------------ launch --

extern "C" void kernel_launch(void* const* d_in, const int* in_sizes, int n_in,
                              void* d_out, int out_size, void* d_ws, size_t ws_size,
                              hipStream_t stream) {
    const float* x        = (const float*)d_in[0];
    const int*   batch    = (const int*)d_in[1];
    const float* e_rank_W = (const float*)d_in[3];
    const float* e_rank_b = (const float*)d_in[4];
    const float* e_card_W = (const float*)d_in[5];
    const float* e_card_b = (const float*)d_in[6];
    const float* e_key_W1 = (const float*)d_in[7];
    const float* e_key_b1 = (const float*)d_in[8];
    const float* e_key_g  = (const float*)d_in[9];
    const float* e_key_be = (const float*)d_in[10];
    const float* e_key_W2 = (const float*)d_in[11];
    const float* e_key_b2 = (const float*)d_in[12];
    const float* e_val_W1 = (const float*)d_in[13];
    const float* e_val_b1 = (const float*)d_in[14];
    const float* e_val_g  = (const float*)d_in[15];
    const float* e_val_be = (const float*)d_in[16];
    const float* e_val_W2 = (const float*)d_in[17];
    const float* e_val_b2 = (const float*)d_in[18];
    const float* d_key_W1 = (const float*)d_in[19];
    const float* d_key_b1 = (const float*)d_in[20];
    const float* d_key_g  = (const float*)d_in[21];
    const float* d_key_be = (const float*)d_in[22];
    const float* d_key_W2 = (const float*)d_in[23];
    const float* d_key_b2 = (const float*)d_in[24];
    const float* d_dec_W1 = (const float*)d_in[25];
    const float* d_dec_b1 = (const float*)d_in[26];
    const float* d_dec_W2 = (const float*)d_in[27];
    const float* d_dec_b2 = (const float*)d_in[28];
    const float* d_size_W1 = (const float*)d_in[29];
    const float* d_size_b1 = (const float*)d_in[30];
    const float* d_size_g  = (const float*)d_in[31];
    const float* d_size_be = (const float*)d_in[32];
    const float* d_size_W2 = (const float*)d_in[33];
    const float* d_size_b2 = (const float*)d_in[34];
    float* out = (float*)d_out;

    // workspace layout
    char* ws = (char*)d_ws;
    size_t off = 0;
    float*  mag    = (float*) (ws + off); off += (size_t)CN * 4;
    int*    cnt    = (int*)   (ws + off); off += (size_t)CB * 4;
    int*    starts = (int*)   (ws + off); off += (size_t)CB * 4;
    int*    rankp  = (int*)   (ws + off); off += (size_t)CN * 4;
    float*  ekey   = (float*) (ws + off); off += (size_t)(CMAXN + 1) * CHID * 4;
    float*  dkey   = (float*) (ws + off); off += (size_t)CMAXN * CHID * 4;
    float*  zbuf   = (float*) (ws + off); off += (size_t)CB * CHID * 4;
    int*    ndec   = (int*)   (ws + off); off += (size_t)CB * 4;
    bf16_t* xb     = (bf16_t*)(ws + off); off += (size_t)CN * CDIM * 2;
    bf16_t* ws_e1  = (bf16_t*)(ws + off); off += (size_t)CDIM * CMIDV * 2;
    bf16_t* ws_e2  = (bf16_t*)(ws + off); off += (size_t)CMIDV * CHID * 2;
    bf16_t* ws_d1  = (bf16_t*)(ws + off); off += (size_t)CHID * CMIDD * 2;
    bf16_t* ws_d2  = (bf16_t*)(ws + off); off += (size_t)CMIDD * CDIM * 2;

    // prep: bf16 x, fragment-swizzled weights
    f32_to_bf16_kernel<<<(CN * CDIM / 4 + 255) / 256, 256, 0, stream>>>(x, xb, CN * CDIM);
    swz_kernel<<<((CMIDV / 16) * (CDIM / 32) * 32 + 255) / 256, 256, 0, stream>>>(e_val_W1, ws_e1, CDIM, CMIDV);
    swz_kernel<<<((CHID / 16) * (CMIDV / 32) * 32 + 255) / 256, 256, 0, stream>>>(e_val_W2, ws_e2, CMIDV, CHID);
    swz_kernel<<<((CMIDD / 16) * (CHID / 32) * 32 + 255) / 256, 256, 0, stream>>>(d_dec_W1, ws_d1, CHID, CMIDD);
    swz_kernel<<<((CDIM / 16) * (CMIDD / 32) * 32 + 255) / 256, 256, 0, stream>>>(d_dec_W2, ws_d2, CMIDD, CDIM);

    zero_i32_kernel<<<(CB + 255) / 256, 256, 0, stream>>>(cnt, CB);
    count_kernel<<<CN / 256, 256, 0, stream>>>(batch, cnt);
    scan_kernel<<<1, 1, 0, stream>>>(cnt, starts);
    mag_kernel<<<CN / 8, 256, 0, stream>>>(x, e_rank_W, e_rank_b, mag);
    rank_kernel<<<CB, 64, 0, stream>>>(cnt, starts, mag, rankp);
    keytab_kernel<<<CMAXN + 1, 256, 0, stream>>>(e_key_W1, e_key_b1, e_key_g, e_key_be,
                                                 e_key_W2, e_key_b2, ekey);
    keytab_kernel<<<CMAXN, 256, 0, stream>>>(d_key_W1, d_key_b1, d_key_g, d_key_be,
                                             d_key_W2, d_key_b2, dkey);
    zinit_kernel<<<CB, 256, 0, stream>>>(cnt, e_card_W, e_card_b, zbuf);
    encoder_mlp_kernel<<<CN / 64, 128, 0, stream>>>(xb, batch, rankp,
        ws_e1, e_val_b1, e_val_g, e_val_be, ws_e2, e_val_b2, ekey, zbuf);
    size_kernel<<<CB, 256, 0, stream>>>(zbuf, d_size_W1, d_size_b1, d_size_g, d_size_be,
                                        d_size_W2, d_size_b2, ndec);
    decoder_mlp_kernel<<<CB, 128, 0, stream>>>(zbuf, dkey, ws_d1, d_dec_b1,
                                               ws_d2, d_dec_b2, ndec, out);
    aux_kernel<<<(CB * CMAXN) / 256, 256, 0, stream>>>(ndec, out);
}
